// AttentionFusion_26379689132116
// MI455X (gfx1250) — compile-verified, matrix operand order checked by reading, not run on hardware
//
#include <hip/hip_runtime.h>
#include <hip/hip_bf16.h>

// ---------------- problem constants ----------------
#define BATCH     8
#define SEQ       1024          // 32*32
#define DMODEL    768
#define CLIPC     1024
#define NHEADS    8
#define HDIM      96
#define INW       16            // clip spatial in
#define OUTW      32            // clip spatial out
#define ROWS      (BATCH*SEQ)   // 8192
#define ATT_SCALE 0.10206207261596575f   // 96^-0.5
#define LOG2E     1.4426950408889634f

typedef _Float16     h8    __attribute__((ext_vector_type(8)));
typedef _Float16     h16   __attribute__((ext_vector_type(16)));
typedef float        f8    __attribute__((ext_vector_type(8)));
typedef unsigned int u32x4 __attribute__((ext_vector_type(4)));
typedef int          i32x4 __attribute__((ext_vector_type(4)));
typedef int          i32x8 __attribute__((ext_vector_type(8)));

#if __has_builtin(__builtin_amdgcn_tensor_load_to_lds) && \
    __has_builtin(__builtin_amdgcn_s_wait_tensorcnt)
#define USE_TDM 1
#else
#define USE_TDM 0
#endif

static __device__ __forceinline__ f8 zero8() {
  f8 z = {0.f,0.f,0.f,0.f,0.f,0.f,0.f,0.f};
  return z;
}

// Build a 16-wide f16 WMMA fragment from two contiguous 8-half (16B) runs,
// per the CDNA5 16-bit A/B VGPR layout:
//   halves 0..7  <- K = hi*8 + 0..7        (first  b128)
//   halves 8..15 <- K = 16 + hi*8 + 0..7   (second b128)
static __device__ __forceinline__ h16 load_frag(const _Float16* p0, const _Float16* p1) {
  h8 lo = *(const h8*)p0;
  h8 hi = *(const h8*)p1;
  h16 r;
#pragma unroll
  for (int i = 0; i < 8; ++i) { r[i] = lo[i]; r[i + 8] = hi[i]; }
  return r;
}

static __device__ __forceinline__ f8 wmma_f16(h16 a, h16 b, f8 c) {
  return __builtin_amdgcn_wmma_f32_16x16x32_f16(false, a, false, b, (short)0, c, false, false);
}

#if USE_TDM
// Issue a TDM 2-D tile load (Global -> LDS), f16 elements, tensor dims == tile
// dims (no OOB clamping needed). Must be executed by exactly one wave with
// uniform arguments; completion via TENSORcnt (in-order per wave).
// D# layout per CDNA5 ISA 8.3/8.4 (group0: count/lds/global/type, group1:
// data_size, tensor_dim0/1, tile_dim0/1/2, dim0-stride).
static __device__ __forceinline__ void tdm_load_tile_f16(
    void* ldsDst, const void* gSrc,
    unsigned dim0 /*elements per row*/, unsigned dim1 /*rows*/,
    unsigned stride0 /*row stride in elements*/) {
  const size_t   ga  = (size_t)gSrc;
  const unsigned lds = (unsigned)(size_t)ldsDst;   // low 32 bits = LDS byte address
  u32x4 g0;
  g0[0] = 1u;                                          // count=1, user descriptor
  g0[1] = lds;                                         // lds_addr
  g0[2] = (unsigned)(ga & 0xFFFFFFFFu);                // global_addr[31:0]
  g0[3] = (unsigned)((ga >> 32) & 0x01FFFFFFu) | (2u << 30);  // addr[56:32] | type=2
  i32x8 g1;
  g1[0] = (int)(1u << 16);                             // data_size = 1 (2 bytes)
  g1[1] = (int)((dim0 & 0xFFFFu) << 16);               // tensor_dim0[15:0]
  g1[2] = (int)((dim0 >> 16) | ((dim1 & 0xFFFFu) << 16));   // dim0 hi | tensor_dim1 lo
  g1[3] = (int)(((dim1 >> 16) & 0xFFFFu) | ((dim0 & 0xFFFFu) << 16)); // dim1 hi | tile_dim0
  g1[4] = (int)(dim1 & 0xFFFFu);                       // tile_dim1 (tile_dim2 = 0)
  g1[5] = (int)stride0;                                // tensor_dim0_stride[31:0]
  g1[6] = 0;                                           // stride0 hi | dim1_stride lo
  g1[7] = 0;
  i32x4 gz = {0, 0, 0, 0};
#if defined(__clang_major__) && (__clang_major__ >= 23)
  i32x8 gz8 = {0, 0, 0, 0, 0, 0, 0, 0};
  __builtin_amdgcn_tensor_load_to_lds(g0, g1, gz, gz, gz8, 0);
#else
  __builtin_amdgcn_tensor_load_to_lds(g0, g1, gz, gz, 0);
#endif
}
#endif

// ---------------- elementwise converter ----------------
__global__ void cvt_f32_to_f16(const float* __restrict__ in, _Float16* __restrict__ out, int n) {
  for (int i = blockIdx.x * blockDim.x + threadIdx.x; i < n; i += gridDim.x * blockDim.x)
    out[i] = (_Float16)in[i];
}

// ---------------- transpose KxN fp32 weight -> NxK f16 (one-time, tiled) -----------------
__global__ __launch_bounds__(256, 1)
void transpose_to_f16(const float* __restrict__ in, _Float16* __restrict__ out, int K, int N) {
  __shared__ float tile[32][33];
  const int tx = threadIdx.x, ty = threadIdx.y;     // block (32, 8)
  const int kb = blockIdx.y * 32, nb = blockIdx.x * 32;
#pragma unroll
  for (int i = 0; i < 4; ++i)
    tile[ty + 8 * i][tx] = in[(size_t)(kb + ty + 8 * i) * N + nb + tx];
  __syncthreads();
#pragma unroll
  for (int i = 0; i < 4; ++i)
    out[(size_t)(nb + ty + 8 * i) * K + kb + tx] = (_Float16)tile[tx][ty + 8 * i];
}

// ---------------- bilinear resize (half-pixel) + pack as GEMM A (ROWS x CLIPC, f16) --------
__global__ void resize_pack(const float* __restrict__ clip, _Float16* __restrict__ Aclip) {
  const int x = blockIdx.x, y = blockIdx.y, b = blockIdx.z;
  const float fy = (y + 0.5f) * 0.5f - 0.5f;
  const float fx = (x + 0.5f) * 0.5f - 0.5f;
  int y0 = (int)floorf(fy); float wy = fy - (float)y0;
  int x0 = (int)floorf(fx); float wx = fx - (float)x0;
  int y0c = min(max(y0, 0), INW - 1), y1c = min(max(y0 + 1, 0), INW - 1);
  int x0c = min(max(x0, 0), INW - 1), x1c = min(max(x0 + 1, 0), INW - 1);
  const size_t outRow = (size_t)(b * SEQ + y * OUTW + x) * CLIPC;
  for (int c = threadIdx.x; c < CLIPC; c += blockDim.x) {
    const float* p = clip + ((size_t)(b * CLIPC + c) * INW) * INW;
    float v00 = p[y0c * INW + x0c], v01 = p[y0c * INW + x1c];
    float v10 = p[y1c * INW + x0c], v11 = p[y1c * INW + x1c];
    float v = (1.f - wy) * ((1.f - wx) * v00 + wx * v01) +
              wy        * ((1.f - wx) * v10 + wx * v11);
    Aclip[outRow + c] = (_Float16)v;
  }
}

// ---------------- generic WMMA GEMM: C(MxN) = A(MxK,f16) * Bt(NxK,f16)^T + bias ------------
// block = 256 threads = 8 waves; block tile 128x64; wave tile 32x32 (2x2 frags); K step 32.
// Both A (128x32) and Bt (64x32) tiles are staged by the Tensor Data Mover,
// double-buffered: wave 0 issues the next k-step's 2 TDM ops, waits TENSORcnt<=2
// (in-order => current buffer landed), then the block computes while DMA runs.
template <bool STORE_HALF>
__global__ __launch_bounds__(256, 1)
void gemm_wmma(const _Float16* __restrict__ A, const _Float16* __restrict__ Bt,
               const float* __restrict__ bias, void* __restrict__ Cout,
               int M, int N, int K) {
  __shared__ __align__(16) _Float16 ldsA[2][128 * 32];
  __shared__ __align__(16) _Float16 ldsBt[2][64 * 32];   // [n][k]

  const int t    = threadIdx.x;
  const int lane = t & 31, w = t >> 5;
  const int hi   = lane >> 4, ln = lane & 15;
  const int mw   = (w >> 1) * 32, nw = (w & 1) * 32;
  const int rowbase = blockIdx.y * 128, colbase = blockIdx.x * 64;

  f8 acc[2][2];
#pragma unroll
  for (int i = 0; i < 2; ++i)
#pragma unroll
    for (int j = 0; j < 2; ++j) acc[i][j] = zero8();

#if USE_TDM
  if (w == 0) {
    tdm_load_tile_f16(&ldsA[0][0],  A  + (size_t)rowbase * K, 32u, 128u, (unsigned)K);
    tdm_load_tile_f16(&ldsBt[0][0], Bt + (size_t)colbase * K, 32u, 64u,  (unsigned)K);
  }
#endif

  int cur = 0;
  for (int k0 = 0; k0 < K; k0 += 32, cur ^= 1) {
#if USE_TDM
    if (w == 0) {
      if (k0 + 32 < K) {
        tdm_load_tile_f16(&ldsA[cur ^ 1][0],  A  + (size_t)rowbase * K + k0 + 32,
                          32u, 128u, (unsigned)K);
        tdm_load_tile_f16(&ldsBt[cur ^ 1][0], Bt + (size_t)colbase * K + k0 + 32,
                          32u, 64u, (unsigned)K);
        __builtin_amdgcn_s_wait_tensorcnt(2);   // current buffer's 2 ops complete
      } else {
        __builtin_amdgcn_s_wait_tensorcnt(0);
      }
    }
#else
    {
      const int rA = t >> 1, koA = (t & 1) * 16;
      const _Float16* g = A + (size_t)(rowbase + rA) * K + k0 + koA;
      *(h8*)&ldsA[cur][rA * 32 + koA]     = *(const h8*)g;
      *(h8*)&ldsA[cur][rA * 32 + koA + 8] = *(const h8*)(g + 8);
      const int nB = t >> 2, koB = (t & 3) * 8;
      *(h8*)&ldsBt[cur][nB * 32 + koB] =
          *(const h8*)(Bt + (size_t)(colbase + nB) * K + k0 + koB);
    }
#endif
    __syncthreads();   // buffer[cur] ready for all waves

    h16 af[2], bf[2];
#pragma unroll
    for (int f = 0; f < 2; ++f) {
      const _Float16* ra = &ldsA[cur][(mw + f * 16 + ln) * 32 + hi * 8];
      af[f] = load_frag(ra, ra + 16);
      const _Float16* rb = &ldsBt[cur][(nw + f * 16 + ln) * 32 + hi * 8];
      bf[f] = load_frag(rb, rb + 16);
    }
#pragma unroll
    for (int mi = 0; mi < 2; ++mi)
#pragma unroll
      for (int ni = 0; ni < 2; ++ni)
        acc[mi][ni] = wmma_f16(af[mi], bf[ni], acc[mi][ni]);
    __syncthreads();   // done reading buffer[cur] before it is re-filled
  }

  // ---- epilogue: C/D layout row = g + 8*hi, col = ln ----
#pragma unroll
  for (int mi = 0; mi < 2; ++mi)
#pragma unroll
    for (int ni = 0; ni < 2; ++ni) {
      const int col = colbase + nw + ni * 16 + ln;
      const float bb = bias[col];
#pragma unroll
      for (int g = 0; g < 8; ++g) {
        const int row = rowbase + mw + mi * 16 + g + hi * 8;
        const float v = acc[mi][ni][g] + bb;
        if (STORE_HALF) ((_Float16*)Cout)[(size_t)row * N + col] = (_Float16)v;
        else            ((float*)Cout)[(size_t)row * N + col] = v;
      }
    }
}

// ---------------- per-row LayerNorm over DMODEL, fp32 in -> f16 or fp32 out ---------------
template <bool OUT_HALF>
__global__ __launch_bounds__(256, 1)
void row_layernorm(const float* __restrict__ in, const float* __restrict__ wgt,
                   const float* __restrict__ bv, void* __restrict__ out, float eps) {
  const int row = blockIdx.x, t = threadIdx.x;
  const float* p = in + (size_t)row * DMODEL;
  float v[3];
  float s = 0.f, q = 0.f;
#pragma unroll
  for (int i = 0; i < 3; ++i) {
    v[i] = p[t + 256 * i];
    s += v[i]; q += v[i] * v[i];
  }
  __shared__ float rs[8], rq[8];
#pragma unroll
  for (int off = 16; off >= 1; off >>= 1) {
    s += __shfl_xor(s, off, 32);
    q += __shfl_xor(q, off, 32);
  }
  if ((t & 31) == 0) { rs[t >> 5] = s; rq[t >> 5] = q; }
  __syncthreads();
  float S = 0.f, Q = 0.f;
#pragma unroll
  for (int i = 0; i < 8; ++i) { S += rs[i]; Q += rq[i]; }
  const float mean = S * (1.f / DMODEL);
  const float var  = Q * (1.f / DMODEL) - mean * mean;
  const float inv  = rsqrtf(var + eps);
#pragma unroll
  for (int i = 0; i < 3; ++i) {
    const int c = t + 256 * i;
    const float o = (v[i] - mean) * inv * wgt[c] + bv[c];
    if (OUT_HALF) ((_Float16*)out)[(size_t)row * DMODEL + c] = (_Float16)o;
    else          ((float*)out)[(size_t)row * DMODEL + c] = o;
  }
}

// ---------------- flash attention: grid (SEQ/64, NHEADS, BATCH), block 128 (4 waves) ------
// Q/K/V stored (ROWS x DMODEL) f16, head slice at column h*96. Each wave owns 16 query rows.
// Q and K tiles staged by the TDM (overlapping the threads' manual V transpose).
__global__ __launch_bounds__(128, 1)
void flash_attn(const _Float16* __restrict__ Q, const _Float16* __restrict__ Kh,
                const _Float16* __restrict__ Vh, _Float16* __restrict__ Oh) {
  __shared__ __align__(16) _Float16 sQ[64 * HDIM];        // 12 KB
  __shared__ __align__(16) _Float16 sK[64 * HDIM];        // 12 KB (row-major keys x c)
  __shared__ __align__(16) _Float16 sVt[HDIM * 64];       // 12 KB (transposed: c x keys)
  __shared__ __align__(16) _Float16 sP[4][16 * 64];       //  8 KB (wave-private P tiles)

  const int t = threadIdx.x, lane = t & 31, w = t >> 5;
  const int hi = lane >> 4, ln = lane & 15;
  const int b = blockIdx.z, hh = blockIdx.y, q0 = blockIdx.x * 64;

  const _Float16* Qg = Q + ((size_t)(b * SEQ + q0)) * DMODEL + hh * HDIM;

  // stage Q tile (64 x 96)
#if USE_TDM
  if (w == 0) {
    tdm_load_tile_f16(sQ, Qg, (unsigned)HDIM, 64u, (unsigned)DMODEL);
    __builtin_amdgcn_s_wait_tensorcnt(0);
  }
#else
#pragma unroll
  for (int i = 0; i < 6; ++i) {
    const int cid = t + 128 * i;
    const int r = cid / 12, co = (cid % 12) * 8;
    *(h8*)&sQ[r * HDIM + co] = *(const h8*)(Qg + (size_t)r * DMODEL + co);
  }
#endif
  __syncthreads();

  // per-wave A fragments of Q (rows w*16 .. w*16+15), hd split 3 x 32
  h16 aq[3];
#pragma unroll
  for (int cc = 0; cc < 3; ++cc) {
    const _Float16* ra = &sQ[(w * 16 + ln) * HDIM + cc * 32 + hi * 8];
    aq[cc] = load_frag(ra, ra + 16);
  }

  float m_run[8], l_run[8];
  f8 o[6];
#pragma unroll
  for (int g = 0; g < 8; ++g) { m_run[g] = -1e30f; l_run[g] = 0.f; }
#pragma unroll
  for (int j = 0; j < 6; ++j) o[j] = zero8();

  for (int kb = 0; kb < SEQ / 64; ++kb) {
    __syncthreads();   // previous iteration done reading sK/sVt
    const _Float16* Kg = Kh + ((size_t)(b * SEQ + kb * 64)) * DMODEL + hh * HDIM;
    const _Float16* Vg = Vh + ((size_t)(b * SEQ + kb * 64)) * DMODEL + hh * HDIM;
#if USE_TDM
    if (w == 0)
      tdm_load_tile_f16(sK, Kg, (unsigned)HDIM, 64u, (unsigned)DMODEL);
#endif
#pragma unroll
    for (int i = 0; i < 6; ++i) {
      const int cid = t + 128 * i;
      const int r = cid / 12, co = (cid % 12) * 8;
#if !USE_TDM
      *(h8*)&sK[r * HDIM + co] = *(const h8*)(Kg + (size_t)r * DMODEL + co);
#endif
      h8 vv = *(const h8*)(Vg + (size_t)r * DMODEL + co);
#pragma unroll
      for (int e = 0; e < 8; ++e) sVt[(co + e) * 64 + r] = vv[e];
    }
#if USE_TDM
    if (w == 0) __builtin_amdgcn_s_wait_tensorcnt(0);
#endif
    __syncthreads();

    // ---- scores: 16 queries x 64 keys = 4 WMMA tiles, hd reduced over 3 chunks ----
    f8 sc[4];
#pragma unroll
    for (int nt = 0; nt < 4; ++nt) {
      sc[nt] = zero8();
#pragma unroll
      for (int cc = 0; cc < 3; ++cc) {
        const _Float16* rb = &sK[(nt * 16 + ln) * HDIM + cc * 32 + hi * 8];
        h16 bf = load_frag(rb, rb + 16);
        sc[nt] = wmma_f16(aq[cc], bf, sc[nt]);
      }
#pragma unroll
      for (int g = 0; g < 8; ++g) sc[nt][g] *= ATT_SCALE;
    }

    // ---- online softmax (row = g + 8*hi; reduce across the 16-lane half) ----
    float tmax[8];
#pragma unroll
    for (int g = 0; g < 8; ++g) {
      float m = sc[0][g];
#pragma unroll
      for (int nt = 1; nt < 4; ++nt) m = fmaxf(m, sc[nt][g]);
#pragma unroll
      for (int off = 8; off >= 1; off >>= 1) m = fmaxf(m, __shfl_xor(m, off, 32));
      tmax[g] = m;
    }
    float corr[8], rsum[8];
#pragma unroll
    for (int g = 0; g < 8; ++g) {
      const float mn = fmaxf(m_run[g], tmax[g]);
      corr[g] = exp2f((m_run[g] - mn) * LOG2E);
      m_run[g] = mn;
      rsum[g] = 0.f;
    }
#pragma unroll
    for (int nt = 0; nt < 4; ++nt)
#pragma unroll
      for (int g = 0; g < 8; ++g) {
        const float p = exp2f((sc[nt][g] - m_run[g]) * LOG2E);
        rsum[g] += p;
        sP[w][(g + 8 * hi) * 64 + nt * 16 + ln] = (_Float16)p;
      }
#pragma unroll
    for (int g = 0; g < 8; ++g) {
      float r = rsum[g];
#pragma unroll
      for (int off = 8; off >= 1; off >>= 1) r += __shfl_xor(r, off, 32);
      l_run[g] = l_run[g] * corr[g] + r;
    }
#pragma unroll
    for (int j = 0; j < 6; ++j)
#pragma unroll
      for (int g = 0; g < 8; ++g) o[j][g] *= corr[g];

    // ---- O += P @ V : keys reduced over 2 chunks of 32, 6 output c-tiles ----
#pragma unroll
    for (int kk = 0; kk < 2; ++kk) {
      const _Float16* rp = &sP[w][ln * 64 + kk * 32 + hi * 8];
      h16 ap = load_frag(rp, rp + 16);
#pragma unroll
      for (int j = 0; j < 6; ++j) {
        const _Float16* rv = &sVt[(j * 16 + ln) * 64 + kk * 32 + hi * 8];
        h16 bv = load_frag(rv, rv + 16);
        o[j] = wmma_f16(ap, bv, o[j]);
      }
    }
  }

  // ---- normalize and scatter into Oh (ROWS x DMODEL f16, head slice) ----
#pragma unroll
  for (int j = 0; j < 6; ++j)
#pragma unroll
    for (int g = 0; g < 8; ++g) {
      const int qi = q0 + w * 16 + g + 8 * hi;
      const float val = o[j][g] / l_run[g];
      Oh[((size_t)(b * SEQ + qi)) * DMODEL + hh * HDIM + j * 16 + ln] = (_Float16)val;
    }
}

// ====================================================================================
extern "C" void kernel_launch(void* const* d_in, const int* in_sizes, int n_in,
                              void* d_out, int out_size, void* d_ws, size_t ws_size,
                              hipStream_t stream) {
  const float* x_f     = (const float*)d_in[0];
  const float* clip    = (const float*)d_in[1];
  const float* conv_w  = (const float*)d_in[2];
  const float* conv_b  = (const float*)d_in[3];
  const float* ln1_w   = (const float*)d_in[4];
  const float* ln1_b   = (const float*)d_in[5];
  const float* wq      = (const float*)d_in[6];
  const float* bq      = (const float*)d_in[7];
  const float* wk      = (const float*)d_in[8];
  const float* bk      = (const float*)d_in[9];
  const float* wv      = (const float*)d_in[10];
  const float* bv      = (const float*)d_in[11];
  const float* wo      = (const float*)d_in[12];
  const float* bo      = (const float*)d_in[13];
  const float* ln2_w   = (const float*)d_in[14];
  const float* ln2_b   = (const float*)d_in[15];
  float* out = (float*)d_out;

  // ---- carve workspace ----
  size_t off = 0;
  auto carve = [&](size_t bytes) -> char* {
    char* p = (char*)d_ws + off;
    off += (bytes + 255) & ~(size_t)255;
    return p;
  };
  _Float16* Wct = (_Float16*)carve((size_t)CLIPC * DMODEL * 2);   // (DMODEL x CLIPC)
  _Float16* Wqt = (_Float16*)carve((size_t)DMODEL * DMODEL * 2);  // (N x K) transposed
  _Float16* Wkt = (_Float16*)carve((size_t)DMODEL * DMODEL * 2);
  _Float16* Wvt = (_Float16*)carve((size_t)DMODEL * DMODEL * 2);
  _Float16* Wot = (_Float16*)carve((size_t)DMODEL * DMODEL * 2);
  _Float16* Xh  = (_Float16*)carve((size_t)ROWS * DMODEL * 2);
  _Float16* Aclip = (_Float16*)carve((size_t)ROWS * CLIPC * 2);   // also reused as Qin
  float*    tmpF = (float*)carve((size_t)ROWS * DMODEL * 4);
  _Float16* Qh  = (_Float16*)carve((size_t)ROWS * DMODEL * 2);
  _Float16* Kh  = (_Float16*)carve((size_t)ROWS * DMODEL * 2);
  _Float16* Vh  = (_Float16*)carve((size_t)ROWS * DMODEL * 2);
  _Float16* Oh  = (_Float16*)carve((size_t)ROWS * DMODEL * 2);
  _Float16* Qin = Aclip;   // reuse (Aclip dead after conv GEMM)

  // ---- 1. weight transpose (KxN fp32 -> NxK f16) and x conversion ----
  transpose_to_f16<<<dim3(DMODEL / 32, CLIPC  / 32), dim3(32, 8), 0, stream>>>(conv_w, Wct, CLIPC,  DMODEL);
  transpose_to_f16<<<dim3(DMODEL / 32, DMODEL / 32), dim3(32, 8), 0, stream>>>(wq,     Wqt, DMODEL, DMODEL);
  transpose_to_f16<<<dim3(DMODEL / 32, DMODEL / 32), dim3(32, 8), 0, stream>>>(wk,     Wkt, DMODEL, DMODEL);
  transpose_to_f16<<<dim3(DMODEL / 32, DMODEL / 32), dim3(32, 8), 0, stream>>>(wv,     Wvt, DMODEL, DMODEL);
  transpose_to_f16<<<dim3(DMODEL / 32, DMODEL / 32), dim3(32, 8), 0, stream>>>(wo,     Wot, DMODEL, DMODEL);
  cvt_f32_to_f16<<<1024, 256, 0, stream>>>(x_f, Xh, ROWS * DMODEL);

  // ---- 2. bilinear resize + pack conv-GEMM A ----
  resize_pack<<<dim3(OUTW, OUTW, BATCH), 256, 0, stream>>>(clip, Aclip);

  // ---- 3. conv 1x1 GEMM: (8192x1024)x(1024x768)+b -> fp32 ----
  gemm_wmma<false><<<dim3(DMODEL / 64, ROWS / 128), 256, 0, stream>>>(
      Aclip, Wct, conv_b, tmpF, ROWS, DMODEL, CLIPC);

  // ---- 4. channel LN (per-row over 768), eps 1e-6 -> f16 q_in ----
  row_layernorm<true><<<ROWS, 256, 0, stream>>>(tmpF, ln1_w, ln1_b, Qin, 1e-6f);

  // ---- 5. Q/K/V projections ----
  gemm_wmma<true><<<dim3(DMODEL / 64, ROWS / 128), 256, 0, stream>>>(
      Qin, Wqt, bq, Qh, ROWS, DMODEL, DMODEL);
  gemm_wmma<true><<<dim3(DMODEL / 64, ROWS / 128), 256, 0, stream>>>(
      Xh, Wkt, bk, Kh, ROWS, DMODEL, DMODEL);
  gemm_wmma<true><<<dim3(DMODEL / 64, ROWS / 128), 256, 0, stream>>>(
      Xh, Wvt, bv, Vh, ROWS, DMODEL, DMODEL);

  // ---- 6. flash attention (WMMA QK^T + online softmax + WMMA PV) ----
  flash_attn<<<dim3(SEQ / 64, NHEADS, BATCH), 128, 0, stream>>>(Qh, Kh, Vh, Oh);

  // ---- 7. output projection -> fp32 ----
  gemm_wmma<false><<<dim3(DMODEL / 64, ROWS / 128), 256, 0, stream>>>(
      Oh, Wot, bo, tmpF, ROWS, DMODEL, DMODEL);

  // ---- 8. final LayerNorm eps 1e-5 -> fp32 out ----
  row_layernorm<false><<<ROWS, 256, 0, stream>>>(tmpF, ln2_w, ln2_b, out, 1e-5f);
}
